// AGNNProp_25391846654263
// MI455X (gfx1250) — matrix operand order repaired and checked by reading, not compiled, not run on hardware
//
#include <hip/hip_runtime.h>

typedef __attribute__((ext_vector_type(16))) _Float16 v16h;
typedef __attribute__((ext_vector_type(8)))  float    v8f;

#define D_FEAT 128

// ---------------------------------------------------------------------------
// Kernel 1: one wave (32 lanes) per node.
//   inv_norm[n] = 1/||x[n]||, denom[n] = 1 (self-loop term, exp(beta-beta)=1),
//   out[n][:]  = x[n][:]      (self-loop contribution, weight 1)
// ---------------------------------------------------------------------------
__global__ void k_node_init(const float* __restrict__ x,
                            float* __restrict__ invn,
                            float* __restrict__ denom,
                            float* __restrict__ out,
                            int n_nodes) {
    int wave = (int)((blockIdx.x * blockDim.x + threadIdx.x) >> 5);
    int lane = threadIdx.x & 31;
    if (wave >= n_nodes) return;
    const float4 v = *(const float4*)(x + (size_t)wave * D_FEAT + lane * 4);
    float s = v.x * v.x + v.y * v.y + v.z * v.z + v.w * v.w;
#pragma unroll
    for (int off = 16; off > 0; off >>= 1)
        s += __shfl_xor(s, off, 32);
    if (lane == 0) {
        invn[wave]  = 1.0f / sqrtf(s);
        denom[wave] = 1.0f;
    }
    *(float4*)(out + (size_t)wave * D_FEAT + lane * 4) = v;
}

// ---------------------------------------------------------------------------
// Kernel 2: one wave per 16-edge tile. Batched edge dot products via WMMA:
//   C = A (16x128 gathered x[row]) x B^T (gathered x[col]) ; alpha_e = diag(C)
//   w[e] = exp(beta*cos - beta);  denom[row[e]] += w[e]  (atomic)
// A layout (16x32 f16, ISA 7.12.2): lane<16 -> M=lane, K in {kb..kb+7, kb+16..kb+23}
// with kb=0; lane>=16 -> kb=8. B layout (32x16): lane<16 -> N=lane, same K as A
// (kb=0); lane>=16 -> N=lane-16, kb=8.
// ---------------------------------------------------------------------------
__global__ void k_edge_alpha_wmma(const float* __restrict__ x,
                                  const int* __restrict__ row,
                                  const int* __restrict__ col,
                                  const float* __restrict__ invn,
                                  const float* __restrict__ beta,
                                  float* __restrict__ w,
                                  float* __restrict__ denom,
                                  int n_edges) {
    int tile = (int)((blockIdx.x * blockDim.x + threadIdx.x) >> 5);
    int lane = threadIdx.x & 31;
    int n_tiles = (n_edges + 15) >> 4;
    if (tile >= n_tiles) return;          // wave-uniform: EXEC all-ones at WMMA

    int e0 = tile << 4;
    int m  = lane & 15;
    int eg = min(e0 + m, n_edges - 1);    // clamped gather index (tail-safe)
    int rA = row[eg];
    int cB = col[eg];

    const float* Ab = x + (size_t)rA * D_FEAT + ((lane < 16) ? 0 : 8);
    const float* Bb = x + (size_t)cB * D_FEAT + ((lane < 16) ? 0 : 8);

    v8f c = {};
#pragma unroll
    for (int kc = 0; kc < D_FEAT; kc += 32) {
        float4 a0 = *(const float4*)(Ab + kc);
        float4 a1 = *(const float4*)(Ab + kc + 4);
        float4 a2 = *(const float4*)(Ab + kc + 16);
        float4 a3 = *(const float4*)(Ab + kc + 20);
        float4 b0 = *(const float4*)(Bb + kc);
        float4 b1 = *(const float4*)(Bb + kc + 4);
        float4 b2 = *(const float4*)(Bb + kc + 16);
        float4 b3 = *(const float4*)(Bb + kc + 20);

        float af[16] = {a0.x, a0.y, a0.z, a0.w, a1.x, a1.y, a1.z, a1.w,
                        a2.x, a2.y, a2.z, a2.w, a3.x, a3.y, a3.z, a3.w};
        float bf[16] = {b0.x, b0.y, b0.z, b0.w, b1.x, b1.y, b1.z, b1.w,
                        b2.x, b2.y, b2.z, b2.w, b3.x, b3.y, b3.z, b3.w};
        v16h a, b;
#pragma unroll
        for (int i = 0; i < 16; ++i) {
            a[i] = (_Float16)af[i];
            b[i] = (_Float16)bf[i];
        }
        c = __builtin_amdgcn_wmma_f32_16x16x32_f16(
                /*neg_a=*/false, a, /*neg_b=*/false, b,
                /*c_mod=*/(short)0, c, /*reuse_a=*/false, /*reuse_b=*/false);
    }

    // Diagonal extraction: m<8 -> lane m, c[m]; m>=8 -> lane m+16, c[m-8]
    bool valid = (lane < 8) || (lane >= 24);
    int  sel   = (lane < 8) ? lane : (lane - 24);
    float d = c[0];
    if (sel == 1) d = c[1];
    if (sel == 2) d = c[2];
    if (sel == 3) d = c[3];
    if (sel == 4) d = c[4];
    if (sel == 5) d = c[5];
    if (sel == 6) d = c[6];
    if (sel == 7) d = c[7];

    int mm = (lane < 8) ? lane : (lane - 16);
    int e  = e0 + mm;
    if (valid && e < n_edges) {
        int r = row[e], cc = col[e];
        float bb    = beta[0];
        float alpha = d * invn[r] * invn[cc] * bb;
        float wv    = __expf(alpha - bb);   // segment max is exactly beta (self loop)
        w[e] = wv;
        unsafeAtomicAdd(&denom[r], wv);
    }
}

// ---------------------------------------------------------------------------
// Kernel 3: one wave per edge: out[row] += w[e] * x[col]   (native f32 atomics)
// ---------------------------------------------------------------------------
__global__ void k_scatter(const float* __restrict__ x,
                          const int* __restrict__ row,
                          const int* __restrict__ col,
                          const float* __restrict__ w,
                          float* __restrict__ out,
                          int n_edges) {
    int e    = (int)((blockIdx.x * blockDim.x + threadIdx.x) >> 5);
    int lane = threadIdx.x & 31;
    if (e >= n_edges) return;
    int   r  = row[e];
    int   cc = col[e];
    float wv = w[e];
    const float4 v = *(const float4*)(x + (size_t)cc * D_FEAT + lane * 4);
    float* o = out + (size_t)r * D_FEAT + lane * 4;
    unsafeAtomicAdd(o + 0, wv * v.x);
    unsafeAtomicAdd(o + 1, wv * v.y);
    unsafeAtomicAdd(o + 2, wv * v.z);
    unsafeAtomicAdd(o + 3, wv * v.w);
}

// ---------------------------------------------------------------------------
// Kernel 4: out[n][f] /= denom[n]
// ---------------------------------------------------------------------------
__global__ void k_scale(float* __restrict__ out,
                        const float* __restrict__ denom,
                        int total) {
    int i = blockIdx.x * blockDim.x + threadIdx.x;
    if (i >= total) return;
    out[i] = out[i] / denom[i >> 7];
}

// ---------------------------------------------------------------------------
extern "C" void kernel_launch(void* const* d_in, const int* in_sizes, int n_in,
                              void* d_out, int out_size, void* d_ws, size_t ws_size,
                              hipStream_t stream) {
    const float* x    = (const float*)d_in[0];
    const float* beta = (const float*)d_in[1];
    const int*   ei   = (const int*)d_in[2];

    const int N = in_sizes[0] / D_FEAT;   // 40000
    const int E = in_sizes[2] / 2;        // 640000
    const int* row = ei;                  // edge_index[0]
    const int* col = ei + E;              // edge_index[1]

    float* out   = (float*)d_out;
    float* invn  = (float*)d_ws;          // N floats
    float* denom = invn + N;              // N floats
    float* w     = denom + N;             // E floats  (total ~2.9 MB scratch)

    // K1: node init (8 waves / 256-thread block)
    int blocks1 = (N + 7) / 8;
    k_node_init<<<blocks1, 256, 0, stream>>>(x, invn, denom, out, N);

    // K2: WMMA edge alphas + denom accumulation
    int tiles   = (E + 15) / 16;
    int blocks2 = (tiles + 7) / 8;
    k_edge_alpha_wmma<<<blocks2, 256, 0, stream>>>(x, row, col, invn, beta, w, denom, E);

    // K3: weighted scatter
    int blocks3 = (E + 7) / 8;
    k_scatter<<<blocks3, 256, 0, stream>>>(x, row, col, w, out, E);

    // K4: normalize
    int total = N * D_FEAT;
    k_scale<<<(total + 255) / 256, 256, 0, stream>>>(out, denom, total);
}
